// RFF_2010044695328
// MI455X (gfx1250) — matrix operand order repaired and checked by reading, not compiled, not run on hardware
//
#include <hip/hip_runtime.h>
#include <hip/hip_bf16.h>
#include <math.h>

#define DIM   128
#define FDIM  2048
#define NROWS 100000
#define EPSV  1e-5f

typedef __attribute__((ext_vector_type(16))) __bf16 v16bf;
typedef __attribute__((ext_vector_type(8)))  float  v8f;

union Frag {
  unsigned int u[8];
  uint4        q[2];
  v16bf        v;
};

// round-to-nearest-even f32 -> bf16 (packed pair into one dword)
__device__ __forceinline__ unsigned int pack_bf16(float lo, float hi) {
  unsigned int a = __float_as_uint(lo);
  unsigned int b = __float_as_uint(hi);
  a = (a + 0x7FFFu + ((a >> 16) & 1u)) >> 16;
  b = (b + 0x7FFFu + ((b >> 16) & 1u)) >> 16;
  return (a & 0xFFFFu) | (b << 16);
}
__device__ __forceinline__ unsigned short f2bf(float f) {
  unsigned int u = __float_as_uint(f);
  return (unsigned short)((u + 0x7FFFu + ((u >> 16) & 1u)) >> 16);
}

// ---------------------------------------------------------------------------
// Kernel 1: Poincare renorm scale per row.  One wave32 per 128-float row:
// each lane loads one float4 (coalesced 512B per wave), butterfly-reduce.
// ---------------------------------------------------------------------------
__global__ __launch_bounds__(256) void rowscale_kernel(
    const float* __restrict__ W, float* __restrict__ scale) {
  const int row  = (blockIdx.x * blockDim.x + threadIdx.x) >> 5;
  const int lane = threadIdx.x & 31;
  if (row >= NROWS) return;
  const float4 v = ((const float4*)(W + (size_t)row * DIM))[lane];
  float s = v.x * v.x + v.y * v.y + v.z * v.z + v.w * v.w;
#pragma unroll
  for (int off = 16; off > 0; off >>= 1) s += __shfl_xor(s, off, 32);
  if (lane == 0) {
    const float norm = sqrtf(s);
    scale[row] = fminf(1.0f, (1.0f - EPSV) / fmaxf(norm, EPSV));
  }
}

// ---------------------------------------------------------------------------
// Kernel 2: transpose+convert Lambdas [128][2048] f32 -> B_T [2048][128] bf16
// via LDS 32x32 tiles so both global reads and writes are coalesced.
// After this, a WMMA B-fragment (one column of Lambdas, two runs of 8 K, 16 apart)
// is two contiguous 16-byte runs.
// ---------------------------------------------------------------------------
__global__ __launch_bounds__(256) void bt_kernel(
    const float* __restrict__ L, unsigned short* __restrict__ Bt) {
  __shared__ float tile[32][33];
  const int n0 = blockIdx.x * 32;   // feature dim (2048)
  const int k0 = blockIdx.y * 32;   // reduction dim (128)
  const int tx = threadIdx.x;       // 0..31
  const int ty = threadIdx.y;       // 0..7
#pragma unroll
  for (int j = 0; j < 4; ++j)
    tile[ty + 8 * j][tx] = L[(size_t)(k0 + ty + 8 * j) * FDIM + n0 + tx];
  __syncthreads();
#pragma unroll
  for (int j = 0; j < 4; ++j)
    Bt[(size_t)(n0 + ty + 8 * j) * DIM + k0 + tx] = f2bf(tile[tx][ty + 8 * j]);
}

// ---------------------------------------------------------------------------
// Kernel 3: fused  out = 0.125 * cos( (s.*W) @ Lambdas + bias )
// Block = 256 threads (8 waves), owns a 32-row M block (2 WMMA M-tiles).
// Wave w owns columns [w*256, w*256+256) = 16 N-tiles of 16.
// A is held in registers as bf16 fragments (reused across all 16 N-tiles),
// B streams from the bf16-transposed copy (L2-resident, 512 KB).
// Per N-tile: one clause of 8x global_load_b128 stages all four B
// fragments into distinct registers, then 8x v_wmma_f32_16x16x32_bf16
// run back-to-back (single s_wait_loadcnt instead of four).
// ---------------------------------------------------------------------------
__global__ __launch_bounds__(256) void rff_wmma_kernel(
    const float* __restrict__ W, const unsigned short* __restrict__ Bt,
    const float* __restrict__ bias, const float* __restrict__ rowscale,
    float* __restrict__ out) {
  const int lane = threadIdx.x & 31;
  const int wv   = threadIdx.x >> 5;   // wave id 0..7
  const int half = lane >> 4;          // lane half selects K sub-blocks
  const int mr   = lane & 15;
  const int m0   = blockIdx.x * 32;

  // --- Build A fragments (16-bit A layout, ISA 7.12.2):
  // lanes 0-15:  M=mr, K = kb*32 + {0..7, 16..23}
  // lanes 16-31: M=mr, K = kb*32 + {8..15, 24..31}
  // i.e. two contiguous 16B chunks at kb*32 + half*8 and +16 more.
  Frag a[2][4];
#pragma unroll
  for (int mt = 0; mt < 2; ++mt) {
    const int m = m0 + mt * 16 + mr;
    const float s = rowscale[m];                       // fold renorm into A
    const float4* Ar = (const float4*)(W + (size_t)m * DIM);
#pragma unroll
    for (int kb = 0; kb < 4; ++kb) {
      const int c = (kb * 32 + half * 8) >> 2;         // float4 index
      const float4 x0 = Ar[c], x1 = Ar[c + 1];         // K rel 0..7
      const float4 y0 = Ar[c + 4], y1 = Ar[c + 5];     // K rel 16..23
      a[mt][kb].u[0] = pack_bf16(s * x0.x, s * x0.y);
      a[mt][kb].u[1] = pack_bf16(s * x0.z, s * x0.w);
      a[mt][kb].u[2] = pack_bf16(s * x1.x, s * x1.y);
      a[mt][kb].u[3] = pack_bf16(s * x1.z, s * x1.w);
      a[mt][kb].u[4] = pack_bf16(s * y0.x, s * y0.y);
      a[mt][kb].u[5] = pack_bf16(s * y0.z, s * y0.w);
      a[mt][kb].u[6] = pack_bf16(s * y1.x, s * y1.y);
      a[mt][kb].u[7] = pack_bf16(s * y1.z, s * y1.w);
    }
  }

  const float OUT_SCALE = 0.125f;  // sqrt(2)/sqrt(128)

#pragma unroll 1
  for (int nt = 0; nt < 16; ++nt) {
    const int n = wv * 256 + nt * 16 + mr;             // this lane's column
    // hint next N-tile's B toward the caches
    if (nt < 15)
      __builtin_prefetch(Bt + (size_t)(n + 16) * DIM + half * 16, 0, 0);

    const float bn = bias[n];

    // Stage all four B fragments (32x16 bf16 B layout: lane's column n,
    // K = kb*32 + half*8 + {0..7, 16..23}, two 16B runs in B_T) in one load clause.
    const unsigned short* bbase = Bt + (size_t)n * DIM + half * 8;
    Frag b[4];
#pragma unroll
    for (int kb = 0; kb < 4; ++kb) {
      const uint4* bp = (const uint4*)(bbase + kb * 32);
      b[kb].q[0] = bp[0];
      b[kb].q[1] = bp[2];
    }

    v8f acc0 = {};
    v8f acc1 = {};
#pragma unroll
    for (int kb = 0; kb < 4; ++kb) {
      acc0 = __builtin_amdgcn_wmma_f32_16x16x32_bf16(
          false, a[0][kb].v, false, b[kb].v, (short)0, acc0, false, false);
      acc1 = __builtin_amdgcn_wmma_f32_16x16x32_bf16(
          false, a[1][kb].v, false, b[kb].v, (short)0, acc1, false, false);
    }

    // Epilogue: C/D layout — VGPR r holds row (r + 8*half), col n.
#pragma unroll
    for (int r = 0; r < 8; ++r) {
      const size_t row0 = (size_t)(m0 + half * 8 + r);
      const size_t row1 = row0 + 16;
      out[row0 * FDIM + n] = OUT_SCALE * __cosf(acc0[r] + bn);
      out[row1 * FDIM + n] = OUT_SCALE * __cosf(acc1[r] + bn);
    }
  }
}

// ---------------------------------------------------------------------------
extern "C" void kernel_launch(void* const* d_in, const int* in_sizes, int n_in,
                              void* d_out, int out_size, void* d_ws, size_t ws_size,
                              hipStream_t stream) {
  const float* W    = (const float*)d_in[0];   // [100000,128] f32
  const float* L    = (const float*)d_in[1];   // [128,2048]   f32
  const float* bias = (const float*)d_in[2];   // [2048]       f32
  float* out        = (float*)d_out;           // [100000,2048] f32

  // workspace layout: rowscale (400 KB) | B_T bf16 (512 KB)
  float* rowscale = (float*)d_ws;
  size_t off = ((size_t)NROWS * sizeof(float) + 255) & ~(size_t)255;
  unsigned short* Bt = (unsigned short*)((char*)d_ws + off);

  rowscale_kernel<<<(NROWS * 32 + 255) / 256, 256, 0, stream>>>(W, rowscale);

  dim3 gt(FDIM / 32, DIM / 32);   // (64, 4)
  dim3 bt(32, 8);
  bt_kernel<<<gt, bt, 0, stream>>>(L, Bt);

  rff_wmma_kernel<<<NROWS / 32, 256, 0, stream>>>(W, Bt, bias, rowscale, out);
}
